// TrackGNN_56315611185727
// MI455X (gfx1250) — matrix operand order repaired and checked by reading, not compiled, not run on hardware
//
#include <hip/hip_runtime.h>
#include <hip/hip_bf16.h>
#include <math.h>

// ---------------------------------------------------------------------------
// CDNA5 (gfx1250) WMMA types & helpers
// ---------------------------------------------------------------------------
typedef __bf16 bf16_t;
typedef __attribute__((ext_vector_type(16))) __bf16 v16bf;
typedef __attribute__((ext_vector_type(8)))  float  v8f;

__device__ __forceinline__ v8f v8f_zero() {
  v8f z;
#pragma unroll
  for (int r = 0; r < 8; ++r) z[r] = 0.0f;
  return z;
}

__device__ __forceinline__ v8f wmma_bf16(v16bf a, v16bf b, v8f c) {
  // D = A(16x32 bf16) * B(32x16 bf16) + C(16x16 f32)
  return __builtin_amdgcn_wmma_f32_16x16x32_bf16(
      /*neg_a=*/false, a, /*neg_b=*/false, b,
      /*c_mod=*/(short)0, c, /*reuse_a=*/false, /*reuse_b=*/false);
}

// Hardware V_TANH_F32 (gfx1250 TRANS op, co-executes with the matrix pipe).
__device__ __forceinline__ float tanh_fast(float x) {
#if __has_builtin(__builtin_amdgcn_tanhf)
  return __builtin_amdgcn_tanhf(x);
#else
  return tanhf(x);
#endif
}
// sigmoid(x) = 0.5 + 0.5*tanh(x/2)  -> 1 TRANS + 2 VALU instead of libm path
__device__ __forceinline__ float sigf(float x) {
  return 0.5f + 0.5f * tanh_fast(0.5f * x);
}

// Build an A fragment (16x32, bf16) from a row-major f32 tile.
// ISA layout: lane%16 = M row; lane>>4 selects K-half group;
// VGPR v<4 -> K = 2v + 8g, v>=4 -> K = 16 + 2(v-4) + 8g; each VGPR = 2 bf16.
__device__ __forceinline__ v16bf load_A_tile(const float* tile, int ld,
                                             int kbase, int kmax) {
  int lane = threadIdx.x & 31;
  int m = lane & 15;
  int g = (lane >> 4) & 1;
  v16bf a;
#pragma unroll
  for (int v = 0; v < 8; ++v) {
    int kb = kbase + ((v < 4) ? (2 * v) : (2 * v + 8)) + 8 * g;
    float f0 = (kb     < kmax) ? tile[m * ld + kb]     : 0.0f;
    float f1 = (kb + 1 < kmax) ? tile[m * ld + kb + 1] : 0.0f;
    a[2 * v]     = (bf16_t)f0;
    a[2 * v + 1] = (bf16_t)f1;
  }
  return a;
}

// A fragment already packed in lane-major bf16 layout (32B per lane) -> 2x b128
__device__ __forceinline__ v16bf load_A_packed(const bf16_t* ph, int kc) {
  int lane = threadIdx.x & 31;
  return *(const v16bf*)(ph + kc * 512 + lane * 16);
}

// B fragments are pre-swizzled (pack_B below): per fragment, lane-major,
// 16 bf16 (32 bytes) contiguous per lane -> two b128 loads.
__device__ __forceinline__ v16bf load_B_frag(const bf16_t* fragbase) {
  int lane = threadIdx.x & 31;
  return *(const v16bf*)(fragbase + (size_t)lane * 16);
}

// ---------------------------------------------------------------------------
// Weight pre-pack: f32 -> fragment-packed bf16 in the ISA B-operand layout.
// B element (K=k, N=n) of fragment (nt, kc): n = nt*16 + (lane&15),
// k = kc*32 + (e<8 ? 8*(lane>>4)+e : 16+8*(lane>>4)+(e-8))  (e = element within lane's 16 bf16).
// transposed=1: W stored (Nout, K) row-major (LSTM weights, used as x@W^T)
// transposed=0: W stored (K, Nout) row-major
// ---------------------------------------------------------------------------
__global__ void pack_B_kernel(const float* __restrict__ W, bf16_t* __restrict__ dst,
                              int K, int Nout, int transposed, int ld) {
  int gid = blockIdx.x * blockDim.x + threadIdx.x;
  int numKC = (K + 31) >> 5;
  int total = (Nout >> 4) * numKC * 32;
  if (gid >= total) return;
  int lane = gid & 31;
  int frag = gid >> 5;
  int kc = frag % numKC;
  int nt = frag / numKC;
  bf16_t* out = dst + (size_t)gid * 16;
  int n = nt * 16 + (lane & 15);
#pragma unroll
  for (int e = 0; e < 16; ++e) {
    int k = kc * 32 + ((e < 8) ? (8 * (lane >> 4) + e) : (16 + 8 * (lane >> 4) + (e - 8)));
    float val = 0.0f;
    if (k < K) val = transposed ? W[(size_t)n * ld + k] : W[(size_t)k * ld + n];
    out[e] = (bf16_t)val;
  }
}

__global__ void combine_bias_kernel(const float* a, const float* b, float* o, int n) {
  int i = blockIdx.x * blockDim.x + threadIdx.x;
  if (i < n) o[i] = a[i] + b[i];
}

// ---------------------------------------------------------------------------
// Fused 2-layer LSTM over 20 timesteps, 16 nodes per block, 4 waves.
// Gate tile assignment: global col-tile gt = wave + 4*tw (tw=0..7), so each
// wave owns i/f/g/o gates of its own 32 hidden columns -> cell update is
// wave-local. h0/h1 live in LDS as *packed bf16 A-fragments* (writer-side
// swizzle), so WMMA operand loads are pure ds_load_b128 with no conversion.
// ---------------------------------------------------------------------------
__global__ void __launch_bounds__(128)
lstm_kernel(const float* __restrict__ seq,
            const bf16_t* __restrict__ Wih0p, const bf16_t* __restrict__ Whh0p,
            const bf16_t* __restrict__ Wih1p, const bf16_t* __restrict__ Whh1p,
            const float* __restrict__ bias0, const float* __restrict__ bias1,
            float* __restrict__ fh) {
  __shared__ float shx[16 * 32];                 // x_t, K padded to 32
  __shared__ __align__(16) bf16_t ph0[4 * 512];  // h0 as 4 packed A fragments
  __shared__ __align__(16) bf16_t ph1[4 * 512];  // h1 as 4 packed A fragments

  int tid = threadIdx.x;
  int m0 = blockIdx.x * 16;
  for (int i = tid; i < 16 * 32; i += 128) shx[i] = 0.0f;
  for (int i = tid; i < 2048; i += 128) {
    ph0[i] = (bf16_t)0.0f;
    ph1[i] = (bf16_t)0.0f;
  }

  int w = tid >> 5;
  int lane = tid & 31;
  int n = lane & 15;
  int g = lane >> 4;

  // Per-lane constants: gate biases and the packed-layout store offset for the
  // two hidden-column blocks this wave owns (b=0: tile w, b=1: tile w+4).
  float bs0[2][4], bs1[2][4];
  int stoff[2];  // element offset into ph*; add r*16 per row
  int jcol[2];
#pragma unroll
  for (int b = 0; b < 2; ++b) {
    int tb = w + 4 * b;
    int j = 16 * tb + n;
    jcol[b] = j;
#pragma unroll
    for (int gate = 0; gate < 4; ++gate) {
      bs0[b][gate] = bias0[gate * 128 + j];
      bs1[b][gate] = bias1[gate * 128 + j];
    }
    // inverse of the A-fragment layout: (row m, col j) -> (frag kc, lane', elem)
    int kc = tb >> 1;
    int kl = ((tb & 1) << 4) | n;  // k within fragment
    int q = kl >> 1, jb = kl & 1;
    int vv, g2;
    if (q < 4)       { vv = q;     g2 = 0; }
    else if (q < 8)  { vv = q - 4; g2 = 1; }
    else if (q < 12) { vv = q - 4; g2 = 0; }
    else             { vv = q - 8; g2 = 1; }
    // lane' = m + 16*g2 with m = r + 8*g ; elem = 2*vv + jb
    stoff[b] = kc * 512 + (8 * g + 16 * g2) * 16 + (2 * vv + jb);
  }

  v8f c0[2], c1[2];
  c0[0] = v8f_zero(); c0[1] = v8f_zero();
  c1[0] = v8f_zero(); c1[1] = v8f_zero();
  __syncthreads();

  for (int t = 0; t < 20; ++t) {
    // stage x_t (16 nodes x 16 dims; cols 16..31 stay zero)
    for (int i = tid; i < 256; i += 128) {
      int m = i >> 4, d = i & 15;
      shx[m * 32 + d] = seq[((size_t)(m0 + m) * 20 + t) * 16 + d];
    }
    __syncthreads();

    // ---- layer 0 gates: G = x_t@Wih0^T + h0@Whh0^T ----
    v8f gacc[8];
    v16bf ax = load_A_tile(shx, 32, 0, 32);
#pragma unroll
    for (int tw = 0; tw < 8; ++tw) {
      int gt = w + 4 * tw;
      v8f acc = wmma_bf16(ax, load_B_frag(Wih0p + (size_t)gt * 512), v8f_zero());
#pragma unroll
      for (int kc = 0; kc < 4; ++kc)
        acc = wmma_bf16(load_A_packed(ph0, kc),
                        load_B_frag(Whh0p + (size_t)(gt * 4 + kc) * 512), acc);
      gacc[tw] = acc;
    }
    __syncthreads();  // all reads of old ph0 complete

    // cell update layer 0 (tw = 2*gate + b) -> new h0 into packed LDS
#pragma unroll
    for (int b = 0; b < 2; ++b) {
#pragma unroll
      for (int r = 0; r < 8; ++r) {
        float iv = gacc[0 + b][r] + bs0[b][0];
        float fv = gacc[2 + b][r] + bs0[b][1];
        float gv = gacc[4 + b][r] + bs0[b][2];
        float ov = gacc[6 + b][r] + bs0[b][3];
        float cc = sigf(fv) * c0[b][r] + sigf(iv) * tanh_fast(gv);
        c0[b][r] = cc;
        ph0[stoff[b] + r * 16] = (bf16_t)(sigf(ov) * tanh_fast(cc));
      }
    }
    __syncthreads();

    // ---- layer 1 gates: G = h0@Wih1^T + h1@Whh1^T ----
#pragma unroll
    for (int tw = 0; tw < 8; ++tw) {
      int gt = w + 4 * tw;
      v8f acc = v8f_zero();
#pragma unroll
      for (int kc = 0; kc < 4; ++kc) {
        acc = wmma_bf16(load_A_packed(ph0, kc),
                        load_B_frag(Wih1p + (size_t)(gt * 4 + kc) * 512), acc);
        acc = wmma_bf16(load_A_packed(ph1, kc),
                        load_B_frag(Whh1p + (size_t)(gt * 4 + kc) * 512), acc);
      }
      gacc[tw] = acc;
    }
    __syncthreads();

#pragma unroll
    for (int b = 0; b < 2; ++b) {
#pragma unroll
      for (int r = 0; r < 8; ++r) {
        float iv = gacc[0 + b][r] + bs1[b][0];
        float fv = gacc[2 + b][r] + bs1[b][1];
        float gv = gacc[4 + b][r] + bs1[b][2];
        float ov = gacc[6 + b][r] + bs1[b][3];
        float cc = sigf(fv) * c1[b][r] + sigf(iv) * tanh_fast(gv);
        c1[b][r] = cc;
        float hh = sigf(ov) * tanh_fast(cc);
        ph1[stoff[b] + r * 16] = (bf16_t)hh;
        if (t == 19)  // final hidden -> global, f32, coalesced over lanes
          fh[(size_t)(m0 + r + 8 * g) * 128 + jcol[b]] = hh;
      }
    }
    __syncthreads();
  }
}

// ---------------------------------------------------------------------------
// Generic WMMA GEMM: OUT(16 rows x 64 cols per block) = act(X @ Bpacked + bias)
// K must be a power of two (64/128/256); kshift = log2(K).
// ---------------------------------------------------------------------------
__global__ void __launch_bounds__(128)
gemm_wmma_kernel(const float* __restrict__ X, const bf16_t* __restrict__ Bp,
                 const float* __restrict__ bias, float* __restrict__ OUT,
                 int K, int kshift, int Nout, int act) {
  __shared__ float shX[16 * 256];  // K <= 256
  int tid = threadIdx.x;
  int m0 = blockIdx.x * 16;
  int total = 16 * K;
  for (int i = tid; i < total; i += 128) {
    int m = i >> kshift, k = i & (K - 1);
    shX[m * K + k] = X[(size_t)(m0 + m) * K + k];
  }
  __syncthreads();

  int w = tid >> 5;
  int gt = blockIdx.y * 4 + w;  // global column tile
  if (gt * 16 < Nout) {
    int numKC = K >> 5;
    v8f acc = v8f_zero();
    for (int kc = 0; kc < numKC; ++kc) {
      v16bf a = load_A_tile(shX, K, kc * 32, K);
      v16bf b = load_B_frag(Bp + (size_t)(gt * numKC + kc) * 512);
      acc = wmma_bf16(a, b, acc);
    }
    int lane = tid & 31;
    int n = lane & 15, g = lane >> 4;
    int col = gt * 16 + n;
    float bv = bias ? bias[col] : 0.0f;
#pragma unroll
    for (int r = 0; r < 8; ++r) {
      float v = acc[r] + bv;
      if (act) v = fmaxf(v, 0.0f);
      OUT[(size_t)(m0 + r + 8 * g) * Nout + col] = v;
    }
  }
}

// ---------------------------------------------------------------------------
// GCN support kernels (degree norm + scatter-add with self-loop).
// unsafeAtomicAdd -> global_atomic_add_f32 (no CAS loop).
// ---------------------------------------------------------------------------
__global__ void deg_init_kernel(float* deg, int N) {
  int i = blockIdx.x * blockDim.x + threadIdx.x;
  if (i < N) deg[i] = 1.0f;  // self-loop
}
__global__ void deg_accum_kernel(const int* __restrict__ dst, float* deg, int E) {
  int e = blockIdx.x * blockDim.x + threadIdx.x;
  if (e < E) unsafeAtomicAdd(&deg[dst[e]], 1.0f);
}
__global__ void dinv_kernel(const float* __restrict__ deg, float* dinv, int N) {
  int i = blockIdx.x * blockDim.x + threadIdx.x;
  if (i < N) dinv[i] = rsqrtf(fmaxf(deg[i], 1.0f));
}
__global__ void acc_init_kernel(const float* __restrict__ y,
                                const float* __restrict__ dinv,
                                float* __restrict__ acc, int N) {
  int idx = blockIdx.x * blockDim.x + threadIdx.x;
  if (idx < N * 128) {
    int i = idx >> 7;
    float di = dinv[i];
    acc[idx] = y[idx] * di * di;  // self-loop contribution
  }
}
__global__ void scatter_kernel(const float* __restrict__ y,
                               const int* __restrict__ src,
                               const int* __restrict__ dst,
                               const float* __restrict__ dinv,
                               float* __restrict__ acc, int E) {
  long long idx = (long long)blockIdx.x * blockDim.x + threadIdx.x;
  long long e = idx >> 5;
  if (e >= E) return;
  int q = ((int)idx & 31) * 4;
  int s = src[e], d = dst[e];
  float nm = dinv[s] * dinv[d];
  float4 v = *(const float4*)(y + (size_t)s * 128 + q);
  float* out = acc + (size_t)d * 128 + q;
  unsafeAtomicAdd(out + 0, v.x * nm);
  unsafeAtomicAdd(out + 1, v.y * nm);
  unsafeAtomicAdd(out + 2, v.z * nm);
  unsafeAtomicAdd(out + 3, v.w * nm);
}
__global__ void bias_relu_kernel(const float* __restrict__ acc,
                                 const float* __restrict__ b,
                                 float* __restrict__ x, int total) {
  int idx = blockIdx.x * blockDim.x + threadIdx.x;
  if (idx < total) x[idx] = fmaxf(acc[idx] + b[idx & 127], 0.0f);
}

// ---------------------------------------------------------------------------
// Edge MLP: 16 edges per block; concat(x[src],x[dst]) -> 128 -> 64 -> sigmoid
// ---------------------------------------------------------------------------
__global__ void __launch_bounds__(128)
edge_mlp_kernel(const float* __restrict__ x,
                const int* __restrict__ src, const int* __restrict__ dst,
                const bf16_t* __restrict__ eW1p, const float* __restrict__ eb1,
                const bf16_t* __restrict__ eW2p, const float* __restrict__ eb2,
                const float* __restrict__ eW3, const float* __restrict__ eb3,
                float* __restrict__ edge_pred) {
  __shared__ float shef[16 * 256];
  __shared__ float shh1[16 * 128];
  __shared__ float shh2[16 * 64];
  int tid = threadIdx.x;
  long long e0 = (long long)blockIdx.x * 16;

  for (int i = tid; i < 16 * 256; i += 128) {
    int m = i >> 8, c = i & 255;
    int node = (c < 128) ? src[e0 + m] : dst[e0 + m];
    shef[i] = x[(size_t)node * 128 + (c & 127)];
  }
  __syncthreads();

  int w = tid >> 5, lane = tid & 31;
  int n = lane & 15, g = lane >> 4;

  // layer 1: 128 cols, K=256
#pragma unroll
  for (int p = 0; p < 2; ++p) {
    int gt = w + 4 * p;
    v8f acc = v8f_zero();
#pragma unroll
    for (int kc = 0; kc < 8; ++kc)
      acc = wmma_bf16(load_A_tile(shef, 256, kc * 32, 256),
                      load_B_frag(eW1p + (size_t)(gt * 8 + kc) * 512), acc);
    int col = gt * 16 + n;
    float bv = eb1[col];
#pragma unroll
    for (int r = 0; r < 8; ++r)
      shh1[(r + 8 * g) * 128 + col] = fmaxf(acc[r] + bv, 0.0f);
  }
  __syncthreads();

  // layer 2: 64 cols, K=128
  {
    int gt = w;
    v8f acc = v8f_zero();
#pragma unroll
    for (int kc = 0; kc < 4; ++kc)
      acc = wmma_bf16(load_A_tile(shh1, 128, kc * 32, 128),
                      load_B_frag(eW2p + (size_t)(gt * 4 + kc) * 512), acc);
    int col = gt * 16 + n;
    float bv = eb2[col];
#pragma unroll
    for (int r = 0; r < 8; ++r)
      shh2[(r + 8 * g) * 64 + col] = fmaxf(acc[r] + bv, 0.0f);
  }
  __syncthreads();

  if (tid < 16) {
    float s = eb3[0];
#pragma unroll
    for (int k = 0; k < 64; ++k) s += shh2[tid * 64 + k] * eW3[k];
    edge_pred[e0 + tid] = sigf(s);
  }
}

// ---------------------------------------------------------------------------
// Final heads: classes (64->4) and softplus uncertainty (64->1)
// ---------------------------------------------------------------------------
__global__ void heads_kernel(const float* __restrict__ hc, const float* __restrict__ hu,
                             const float* __restrict__ cW2, const float* __restrict__ cb2,
                             const float* __restrict__ uW2, const float* __restrict__ ub2,
                             float* __restrict__ classes, float* __restrict__ unc, int N) {
  int i = blockIdx.x * blockDim.x + threadIdx.x;
  if (i >= N) return;
  float o[4] = {cb2[0], cb2[1], cb2[2], cb2[3]};
#pragma unroll 4
  for (int k = 0; k < 64; ++k) {
    float h = hc[(size_t)i * 64 + k];
#pragma unroll
    for (int c = 0; c < 4; ++c) o[c] += h * cW2[k * 4 + c];
  }
#pragma unroll
  for (int c = 0; c < 4; ++c) classes[(size_t)i * 4 + c] = o[c];
  float s = ub2[0];
#pragma unroll 4
  for (int k = 0; k < 64; ++k) s += hu[(size_t)i * 64 + k] * uW2[k];
  unc[i] = (s > 20.0f) ? s : log1pf(__expf(s));
}

// ---------------------------------------------------------------------------
// Host orchestration
// ---------------------------------------------------------------------------
extern "C" void kernel_launch(void* const* d_in, const int* in_sizes, int n_in,
                              void* d_out, int out_size, void* d_ws, size_t ws_size,
                              hipStream_t stream) {
  (void)n_in; (void)out_size; (void)ws_size;
  const int N = in_sizes[0] / (20 * 16);  // 50000
  const int E = in_sizes[1] / 2;          // 800000

  const float* seq  = (const float*)d_in[0];
  const int*   eidx = (const int*)d_in[1];
  const int*   esrc = eidx;
  const int*   edst = eidx + E;

  // workspace carve-out
  char* ws = (char*)d_ws;
  size_t off = 0;
  auto carve = [&](size_t bytes) -> void* {
    void* p = (void*)(ws + off);
    off += (bytes + 255) & ~(size_t)255;
    return p;
  };
  bf16_t* Wih0p = (bf16_t*)carve(16384 * 2);
  bf16_t* Whh0p = (bf16_t*)carve(65536 * 2);
  bf16_t* Wih1p = (bf16_t*)carve(65536 * 2);
  bf16_t* Whh1p = (bf16_t*)carve(65536 * 2);
  bf16_t* enc1p = (bf16_t*)carve(16384 * 2);
  bf16_t* enc2p = (bf16_t*)carve(8192 * 2);
  bf16_t* g0p   = (bf16_t*)carve(8192 * 2);
  bf16_t* g1p   = (bf16_t*)carve(16384 * 2);
  bf16_t* g2p   = (bf16_t*)carve(16384 * 2);
  bf16_t* e1p   = (bf16_t*)carve(32768 * 2);
  bf16_t* e2p   = (bf16_t*)carve(8192 * 2);
  bf16_t* c1p   = (bf16_t*)carve(8192 * 2);
  bf16_t* u1p   = (bf16_t*)carve(8192 * 2);
  float* bias0  = (float*)carve(512 * 4);
  float* bias1  = (float*)carve(512 * 4);
  float* bufA   = (float*)carve((size_t)N * 128 * 4);
  float* bufB   = (float*)carve((size_t)N * 128 * 4);
  float* bufC   = (float*)carve((size_t)N * 128 * 4);
  float* bufD   = (float*)carve((size_t)N * 64 * 4);
  float* deg    = (float*)carve((size_t)N * 4);
  float* dinv   = (float*)carve((size_t)N * 4);

  float* out_x       = (float*)d_out;                 // N*128
  float* out_edge    = out_x + (size_t)N * 128;       // E
  float* out_classes = out_edge + (size_t)E;          // N*4
  float* out_unc     = out_classes + (size_t)N * 4;   // N

  auto pack = [&](const void* W, bf16_t* dst, int K, int Nout, int tr, int ld) {
    int numKC = (K + 31) >> 5;
    int total = (Nout >> 4) * numKC * 32;
    pack_B_kernel<<<(total + 127) / 128, 128, 0, stream>>>(
        (const float*)W, dst, K, Nout, tr, ld);
  };
  pack(d_in[2],  Wih0p, 16,  512, 1, 16);
  pack(d_in[3],  Whh0p, 128, 512, 1, 128);
  pack(d_in[6],  Wih1p, 128, 512, 1, 128);
  pack(d_in[7],  Whh1p, 128, 512, 1, 128);
  pack(d_in[10], enc1p, 128, 128, 0, 128);
  pack(d_in[12], enc2p, 128, 64,  0, 64);
  pack(d_in[14], g0p,   64,  128, 0, 128);
  pack(d_in[16], g1p,   128, 128, 0, 128);
  pack(d_in[18], g2p,   128, 128, 0, 128);
  pack(d_in[20], e1p,   256, 128, 0, 128);
  pack(d_in[22], e2p,   128, 64,  0, 64);
  pack(d_in[26], c1p,   128, 64,  0, 64);
  pack(d_in[30], u1p,   128, 64,  0, 64);

  combine_bias_kernel<<<2, 256, 0, stream>>>((const float*)d_in[4], (const float*)d_in[5], bias0, 512);
  combine_bias_kernel<<<2, 256, 0, stream>>>((const float*)d_in[8], (const float*)d_in[9], bias1, 512);

  // fused 2-layer LSTM -> final hidden (bufA)
  lstm_kernel<<<N / 16, 128, 0, stream>>>(seq, Wih0p, Whh0p, Wih1p, Whh1p,
                                          bias0, bias1, bufA);

  // encoder: bufB = relu(fh @ encW1 + b1); bufD(latent) = bufB @ encW2 + b2
  gemm_wmma_kernel<<<dim3(N / 16, 2), 128, 0, stream>>>(bufA, enc1p, (const float*)d_in[11], bufB, 128, 7, 128, 1);
  gemm_wmma_kernel<<<dim3(N / 16, 1), 128, 0, stream>>>(bufB, enc2p, (const float*)d_in[13], bufD, 128, 7, 64, 0);

  // degree norm (once)
  deg_init_kernel<<<(N + 255) / 256, 256, 0, stream>>>(deg, N);
  deg_accum_kernel<<<(E + 255) / 256, 256, 0, stream>>>(edst, deg, E);
  dinv_kernel<<<(N + 255) / 256, 256, 0, stream>>>(deg, dinv, N);

  const bf16_t* gp[3]   = {g0p, g1p, g2p};
  const float* gb[3]    = {(const float*)d_in[15], (const float*)d_in[17], (const float*)d_in[19]};
  const float* xin[3]   = {bufD, bufA, bufB};
  float*       ybuf[3]  = {bufB, bufB, bufA};
  float*       xout3[3] = {bufA, bufB, out_x};
  const int    Kin[3]   = {64, 128, 128};
  const int    Ksh[3]   = {6, 7, 7};

  int sc_blocks = (int)(((long long)E * 32 + 255) / 256);
  for (int l = 0; l < 3; ++l) {
    gemm_wmma_kernel<<<dim3(N / 16, 2), 128, 0, stream>>>(xin[l], gp[l], nullptr, ybuf[l], Kin[l], Ksh[l], 128, 0);
    acc_init_kernel<<<(N * 128 + 255) / 256, 256, 0, stream>>>(ybuf[l], dinv, bufC, N);
    scatter_kernel<<<sc_blocks, 256, 0, stream>>>(ybuf[l], esrc, edst, dinv, bufC, E);
    bias_relu_kernel<<<(N * 128 + 255) / 256, 256, 0, stream>>>(bufC, gb[l], xout3[l], N * 128);
  }

  // heads: hc (bufD) and hu (bufA reused; GCN layer-2 y is dead after scatter)
  float* hu = bufA;
  gemm_wmma_kernel<<<dim3(N / 16, 1), 128, 0, stream>>>(out_x, c1p, (const float*)d_in[27], bufD, 128, 7, 64, 1);
  gemm_wmma_kernel<<<dim3(N / 16, 1), 128, 0, stream>>>(out_x, u1p, (const float*)d_in[31], hu,   128, 7, 64, 1);
  heads_kernel<<<(N + 255) / 256, 256, 0, stream>>>(bufD, hu,
      (const float*)d_in[28], (const float*)d_in[29],
      (const float*)d_in[32], (const float*)d_in[33],
      out_classes, out_unc, N);

  // edge MLP
  edge_mlp_kernel<<<E / 16, 128, 0, stream>>>(out_x, esrc, edst,
      e1p, (const float*)d_in[21], e2p, (const float*)d_in[23],
      (const float*)d_in[24], (const float*)d_in[25], out_edge);
}
